// SwinBlock_87797721465069
// MI455X (gfx1250) — compile-verified, matrix operand order checked by reading, not run on hardware
//
#include <hip/hip_runtime.h>

// ---------------------------------------------------------------------------
// Swin block on gfx1250 (MI455X): all GEMMs via v_wmma_f32_16x16x32_f16,
// f16 activations/weights with f32 accumulate, fused epilogues.
// A-tiles staged to LDS with GLOBAL_LOAD_ASYNC_TO_LDS_B128 (double-buffered,
// s_wait_asynccnt) when the toolchain exposes the builtin.
// ---------------------------------------------------------------------------

typedef __attribute__((ext_vector_type(16))) _Float16 v16h;
typedef __attribute__((ext_vector_type(8)))  _Float16 v8h;
typedef __attribute__((ext_vector_type(8)))  float    v8f;
typedef __attribute__((ext_vector_type(4)))  int      v4i;

#define R_TOK 131072   // 32 images * 4096 tokens
#define DIMC  256
#define NWIN  2048     // 32 * 8 * 8 windows
#define HDIM  32
#define NHEAD 8

#if defined(__has_builtin)
#if __has_builtin(__builtin_amdgcn_global_load_async_to_lds_b128) && \
    __has_builtin(__builtin_amdgcn_s_wait_asynccnt)
#define USE_ASYNC_LDS 1
#endif
#endif
#ifndef USE_ASYNC_LDS
#define USE_ASYNC_LDS 0
#endif

// builtin parameter types: (int4 AS1* src, int4 AS3* dst, imm offset, imm cpol)
typedef __attribute__((address_space(1))) v4i* gptr_v4i;
typedef __attribute__((address_space(3))) v4i* lptr_v4i;

__device__ __forceinline__ v8f wmma16(v16h a, v16h b, v8f c) {
  // D = A(16x32 f16) * B(32x16 f16) + C(16x16 f32)
  return __builtin_amdgcn_wmma_f32_16x16x32_f16(
      /*neg_a=*/false, a, /*neg_b=*/false, b,
      /*c_mod=*/(short)0, c, /*reuse_a=*/false, /*reuse_b=*/false);
}

// Load one lane's A/B fragment for 16-bit 16x32 WMMA operands from a
// row-major source: elements 0..7 at p[0..7], elements 8..15 at p[16..23].
__device__ __forceinline__ v16h load_frag16(const _Float16* p) {
  v8h lo = *(const v8h*)p;
  v8h hi = *(const v8h*)(p + 16);
  v16h r;
#pragma unroll
  for (int j = 0; j < 8; ++j) { r[j] = lo[j]; r[j + 8] = hi[j]; }
  return r;
}

// ---------------------------------------------------------------------------
// Weight repack: f32 row-major (K x N)  ->  f16 B-fragment layout
// [N/16][K/32][lane 0..31][16 halves]; one thread emits one lane vector.
// ---------------------------------------------------------------------------
__global__ void pack_b_kernel(const float* __restrict__ w, _Float16* __restrict__ out,
                              int K, int N) {
  int tid = blockIdx.x * blockDim.x + threadIdx.x;
  int ksteps = K >> 5;
  int total = (N >> 4) * ksteps * 32;
  if (tid >= total) return;
  int lane = tid & 31;
  int rem  = tid >> 5;
  int ks   = rem % ksteps;
  int nt   = rem / ksteps;
  int g = lane >> 4, li = lane & 15;
  int n = nt * 16 + li;
  _Float16* dst = out + (size_t)tid * 16;
#pragma unroll
  for (int j = 0; j < 16; ++j) {
    int k = ks * 32 + ((j < 8) ? (g * 8 + j) : (16 + g * 8 + (j - 8)));
    dst[j] = (_Float16)w[(size_t)k * N + n];
  }
}

// ---------------------------------------------------------------------------
// LayerNorm over 256 channels; one block per token row.
// PERM=true additionally applies the Swin window-partition permutation
// ('b (ws1 h2) (ws2 w2) c -> (b h2 w2) (ws1 ws2) c') to the output row index.
// ---------------------------------------------------------------------------
template <bool PERM>
__global__ void ln_kernel(const float* __restrict__ x,
                          const float* __restrict__ gma,
                          const float* __restrict__ bta,
                          _Float16* __restrict__ y) {
  long row = blockIdx.x;
  int c = threadIdx.x;                 // 256 threads
  float v = x[row * DIMC + c];
  __shared__ float red[DIMC];
  __shared__ float mv[2];
  red[c] = v;
  __syncthreads();
  for (int s = 128; s > 0; s >>= 1) {
    if (c < s) red[c] += red[c + s];
    __syncthreads();
  }
  if (c == 0) mv[0] = red[0] * (1.0f / DIMC);
  __syncthreads();
  float d = v - mv[0];
  red[c] = d * d;
  __syncthreads();
  for (int s = 128; s > 0; s >>= 1) {
    if (c < s) red[c] += red[c + s];
    __syncthreads();
  }
  if (c == 0) mv[1] = rsqrtf(red[0] * (1.0f / DIMC) + 1e-5f);
  __syncthreads();
  float o = d * mv[1] * gma[c] + bta[c];

  long orow = row;
  if (PERM) {
    int b = (int)(row >> 12);          // /4096
    int n = (int)(row & 4095);
    int r = n >> 6, col = n & 63;
    int ws1 = r >> 3, h2i = r & 7, ws2 = col >> 3, w2i = col & 7;
    int win = (b * 8 + h2i) * 8 + w2i;
    orow = (long)win * 64 + ws1 * 8 + ws2;
  }
  y[orow * DIMC + c] = (_Float16)o;
}

// ---------------------------------------------------------------------------
// WMMA GEMM: C(M x N) = A(M x K, f16 row-major) * Bpack + bias, fused epilogue.
// Block = 256 threads = 8 waves; wave tile 32(M) x 64(N) (8 f32 accumulators);
// block tile 64(M) x 256(N). Grid = (M/64, N/256).
// The 64x32 A tile per k-step is staged to LDS (async copy, double buffered)
// and shared by all 8 waves; B fragments stream from L2-resident packed
// weights as single 32B vector loads.
// ---------------------------------------------------------------------------
#define EPI_QKV  0
#define EPI_PROJ 1
#define EPI_GELU 2
#define EPI_OUT  3

template <int N, int K, int EPI>
__global__ void gemm_wmma(const _Float16* __restrict__ A,
                          const _Float16* __restrict__ Bp,
                          const float* __restrict__ bias,
                          _Float16* __restrict__ o0,   // QKV: q  | GELU: hidden
                          _Float16* __restrict__ o1,   // QKV: k
                          _Float16* __restrict__ o2,   // QKV: v
                          float* __restrict__ of,      // PROJ: x1 | OUT: d_out
                          const float* __restrict__ rf // PROJ: x  | OUT: x1
                          ) {
  constexpr int KSTEPS = K / 32;
  __shared__ _Float16 asA[2][64 * 32];                 // 2 x 4KB A tiles

  int lane = threadIdx.x & 31;
  int w    = threadIdx.x >> 5;
  int mw = w & 1, nw = w >> 1;
  int g = lane >> 4, li = lane & 15;
  long mblk = (long)blockIdx.x * 64;
  int  n0   = blockIdx.y * 256 + nw * 64;

  // staging assignment: thread -> one 16B chunk of the 64x32 tile
  int srow = threadIdx.x >> 2;              // 0..63
  int scol = (threadIdx.x & 3) * 8;         // halves 0,8,16,24
  const _Float16* sbase = A + (mblk + srow) * K + scol;
  _Float16* sdst0 = &asA[0][srow * 32 + scol];
  _Float16* sdst1 = &asA[1][srow * 32 + scol];

  auto stage = [&](int ks) {
    const _Float16* src = sbase + ks * 32;
    _Float16* dst = (ks & 1) ? sdst1 : sdst0;
#if USE_ASYNC_LDS
    __builtin_amdgcn_global_load_async_to_lds_b128(
        (gptr_v4i)const_cast<_Float16*>(src), (lptr_v4i)dst, 0, 0);
#else
    *(v8h*)dst = *(const v8h*)src;
#endif
  };

  v8f acc[2][4];
#pragma unroll
  for (int mi = 0; mi < 2; ++mi)
#pragma unroll
    for (int nt = 0; nt < 4; ++nt)
#pragma unroll
      for (int i = 0; i < 8; ++i) acc[mi][nt][i] = 0.0f;

  stage(0);

#pragma unroll
  for (int ks = 0; ks < KSTEPS; ++ks) {
#if USE_ASYNC_LDS
    __builtin_amdgcn_s_wait_asynccnt(0);
#endif
    __syncthreads();
    if (ks + 1 < KSTEPS) stage(ks + 1);    // overlaps with compute below

    const _Float16* at = asA[ks & 1];
    v16h af[2];
#pragma unroll
    for (int mi = 0; mi < 2; ++mi)
      af[mi] = load_frag16(at + (mw * 32 + mi * 16 + li) * 32 + g * 8);

#pragma unroll
    for (int nt = 0; nt < 4; ++nt) {
      const v16h bf = *(const v16h*)(Bp +
          ((((size_t)(n0 >> 4) + nt) * KSTEPS + ks) * 32 + lane) * 16);
#pragma unroll
      for (int mi = 0; mi < 2; ++mi)
        acc[mi][nt] = wmma16(af[mi], bf, acc[mi][nt]);
    }
  }

  // Epilogue: C layout -> lane holds (m = r + 8*g, n = nt*16 + li)
#pragma unroll
  for (int mi = 0; mi < 2; ++mi) {
#pragma unroll
    for (int nt = 0; nt < 4; ++nt) {
#pragma unroll
      for (int r = 0; r < 8; ++r) {
        long m = mblk + mw * 32 + mi * 16 + r + 8 * g;
        int  n = n0 + nt * 16 + li;
        float val = acc[mi][nt][r] + bias[n];

        if (EPI == EPI_QKV) {
          // qkv_w cols ordered (head, d, {q,k,v}); rows window-order tokens
          int s = n % 3, hd = n / 3;
          int hh = hd >> 5, d = hd & 31;
          int win = (int)(m >> 6), tokw = (int)(m & 63);
          if (s == 0) val *= 0.17677669529663687f;  // HD^-0.5
          _Float16* dst = (s == 0) ? o0 : (s == 1) ? o1 : o2;
          dst[(((size_t)win * NHEAD + hh) * 64 + tokw) * HDIM + d] = (_Float16)val;
        } else if (EPI == EPI_PROJ) {
          // window-reverse + residual with original x; write f32 x1
          int win = (int)(m >> 6), tokw = (int)(m & 63);
          int b = win >> 6, h2i = (win >> 3) & 7, w2i = win & 7;
          int ws1 = tokw >> 3, ws2 = tokw & 7;
          int nr = (ws1 * 8 + h2i) * 64 + (ws2 * 8 + w2i);
          size_t idx = ((size_t)b * 4096 + nr) * DIMC + n;
          of[idx] = val + rf[idx];
        } else if (EPI == EPI_GELU) {
          val = 0.5f * val * (1.0f + erff(val * 0.7071067811865475f));
          o0[m * N + n] = (_Float16)val;
        } else { // EPI_OUT
          size_t idx = (size_t)m * DIMC + n;
          of[idx] = val + rf[idx];
        }
      }
    }
  }
}

// ---------------------------------------------------------------------------
// Windowed attention: one block per window, one wave (32 lanes) per head.
// S = q*k^T is one WMMA per 16x16 tile (K = 32 = head_dim); softmax uses
// level-outer batched shuffles (8 independent ds_bpermute per level so the
// dscnt waits coalesce); P re-layout through wave-private LDS; out = P*v.
// ---------------------------------------------------------------------------
__global__ void attn_kernel(const _Float16* __restrict__ q,
                            const _Float16* __restrict__ k,
                            const _Float16* __restrict__ v,
                            _Float16* __restrict__ ao) {
  __shared__ _Float16 pbuf[NHEAD * 16 * 64];   // 16 KB, per-wave 16x64 tile
  int win  = blockIdx.x;
  int h    = threadIdx.x >> 5;
  int lane = threadIdx.x & 31;
  int g = lane >> 4, li = lane & 15;

  const _Float16* qh = q + ((size_t)win * NHEAD + h) * 64 * HDIM;
  const _Float16* kh = k + ((size_t)win * NHEAD + h) * 64 * HDIM;
  const _Float16* vh = v + ((size_t)win * NHEAD + h) * 64 * HDIM;

  // k^T as B fragments (4 n-tiles): B[kk][n] = k[n][kk] -> contiguous rows of k
  v16h kf[4];
#pragma unroll
  for (int nt = 0; nt < 4; ++nt)
    kf[nt] = load_frag16(kh + (nt * 16 + li) * HDIM + g * 8);

  // v as B fragments for P@v (K = 64 -> 2 k-steps, 2 n-tiles of head_dim 32)
  v16h vf[2][2];
#pragma unroll
  for (int ks = 0; ks < 2; ++ks)
#pragma unroll
    for (int nt = 0; nt < 2; ++nt)
#pragma unroll
      for (int j = 0; j < 16; ++j) {
        int kk = ks * 32 + ((j < 8) ? (g * 8 + j) : (16 + g * 8 + (j - 8)));
        vf[ks][nt][j] = vh[kk * HDIM + nt * 16 + li];
      }

  _Float16* pw = pbuf + h * (16 * 64);

  for (int mt = 0; mt < 4; ++mt) {
    int r0 = mt * 16;
    v16h qf = load_frag16(qh + (r0 + li) * HDIM + g * 8);

    v8f S[4];
#pragma unroll
    for (int nt = 0; nt < 4; ++nt) {
      v8f z;
#pragma unroll
      for (int i = 0; i < 8; ++i) z[i] = 0.0f;
      S[nt] = wmma16(qf, kf[nt], z);
    }

    // --- row softmax, level-outer so shuffles per level are independent ---
    float mx[8], sm[8];
#pragma unroll
    for (int r = 0; r < 8; ++r)
      mx[r] = fmaxf(fmaxf(S[0][r], S[1][r]), fmaxf(S[2][r], S[3][r]));
#pragma unroll
    for (int off = 1; off < 16; off <<= 1) {
      float t[8];
#pragma unroll
      for (int r = 0; r < 8; ++r) t[r] = __shfl_xor(mx[r], off, 32);
#pragma unroll
      for (int r = 0; r < 8; ++r) mx[r] = fmaxf(mx[r], t[r]);
    }
#pragma unroll
    for (int r = 0; r < 8; ++r) {
      float s = 0.0f;
#pragma unroll
      for (int nt = 0; nt < 4; ++nt) {
        float p = __expf(S[nt][r] - mx[r]);
        S[nt][r] = p;
        s += p;
      }
      sm[r] = s;
    }
#pragma unroll
    for (int off = 1; off < 16; off <<= 1) {
      float t[8];
#pragma unroll
      for (int r = 0; r < 8; ++r) t[r] = __shfl_xor(sm[r], off, 32);
#pragma unroll
      for (int r = 0; r < 8; ++r) sm[r] += t[r];
    }
#pragma unroll
    for (int r = 0; r < 8; ++r) {
      float inv = 1.0f / sm[r];
#pragma unroll
      for (int nt = 0; nt < 4; ++nt) S[nt][r] *= inv;
    }

    // C-layout -> A-fragment layout via wave-private LDS (ds ops, no barrier)
#pragma unroll
    for (int nt = 0; nt < 4; ++nt)
#pragma unroll
      for (int r = 0; r < 8; ++r)
        pw[(r + 8 * g) * 64 + nt * 16 + li] = (_Float16)S[nt][r];

    v8f oc[2];
#pragma unroll
    for (int nt = 0; nt < 2; ++nt)
#pragma unroll
      for (int i = 0; i < 8; ++i) oc[nt][i] = 0.0f;
#pragma unroll
    for (int ks = 0; ks < 2; ++ks) {
      v16h pf = load_frag16(pw + li * 64 + ks * 32 + g * 8);
#pragma unroll
      for (int nt = 0; nt < 2; ++nt)
        oc[nt] = wmma16(pf, vf[ks][nt], oc[nt]);
    }

    // store per-head output into (window-order token) x 256 activation matrix
#pragma unroll
    for (int nt = 0; nt < 2; ++nt)
#pragma unroll
      for (int r = 0; r < 8; ++r) {
        size_t row = (size_t)win * 64 + r0 + r + 8 * g;
        int col = h * HDIM + nt * 16 + li;
        ao[row * DIMC + col] = (_Float16)oc[nt][r];
      }
  }
}

// ---------------------------------------------------------------------------
extern "C" void kernel_launch(void* const* d_in, const int* in_sizes, int n_in,
                              void* d_out, int out_size, void* d_ws, size_t ws_size,
                              hipStream_t stream) {
  const float* x      = (const float*)d_in[0];
  const float* ln1_g  = (const float*)d_in[1];
  const float* ln1_b  = (const float*)d_in[2];
  const float* qkv_w  = (const float*)d_in[3];
  const float* qkv_b  = (const float*)d_in[4];
  const float* proj_w = (const float*)d_in[5];
  const float* proj_b = (const float*)d_in[6];
  const float* ln2_g  = (const float*)d_in[7];
  const float* ln2_b  = (const float*)d_in[8];
  const float* mlp_w1 = (const float*)d_in[9];
  const float* mlp_b1 = (const float*)d_in[10];
  const float* mlp_w2 = (const float*)d_in[11];
  const float* mlp_b2 = (const float*)d_in[12];

  // ---- workspace layout (halves unless noted) ----
  _Float16* wqkv_p = (_Float16*)d_ws;                 // 256*768
  _Float16* wproj_p = wqkv_p + 256 * 768;             // 256*256
  _Float16* w1_p    = wproj_p + 256 * 256;            // 256*1024
  _Float16* w2_p    = w1_p + 256 * 1024;              // 1024*256
  size_t woff = ((size_t)(256*768 + 256*256 + 256*1024 + 1024*256) * 2 + 255) & ~(size_t)255;
  _Float16* yln = (_Float16*)((char*)d_ws + woff);    // R_TOK*256 (LN1 / LN2)
  _Float16* qb  = yln + (size_t)R_TOK * 256;
  _Float16* kb  = qb  + (size_t)R_TOK * 256;
  _Float16* vb  = kb  + (size_t)R_TOK * 256;
  _Float16* ao  = vb  + (size_t)R_TOK * 256;
  float*    x1  = (float*)(ao + (size_t)R_TOK * 256); // R_TOK*256 f32
  _Float16* mh  = qb;  // MLP hidden (R_TOK*1024) exactly overlays q|k|v|ao

  // ---- weight repack to WMMA B-fragment layout ----
  {
    int t;
    t = (768 / 16) * (256 / 32) * 32;
    pack_b_kernel<<<(t + 255) / 256, 256, 0, stream>>>(qkv_w, wqkv_p, 256, 768);
    t = (256 / 16) * (256 / 32) * 32;
    pack_b_kernel<<<(t + 255) / 256, 256, 0, stream>>>(proj_w, wproj_p, 256, 256);
    t = (1024 / 16) * (256 / 32) * 32;
    pack_b_kernel<<<(t + 255) / 256, 256, 0, stream>>>(mlp_w1, w1_p, 256, 1024);
    t = (256 / 16) * (1024 / 32) * 32;
    pack_b_kernel<<<(t + 255) / 256, 256, 0, stream>>>(mlp_w2, w2_p, 1024, 256);
  }

  // ---- LN1 fused with window partition ----
  ln_kernel<true><<<R_TOK, 256, 0, stream>>>(x, ln1_g, ln1_b, yln);

  // ---- QKV GEMM (131072 x 768, K=256) ----
  gemm_wmma<768, 256, EPI_QKV><<<dim3(R_TOK / 64, 3), 256, 0, stream>>>(
      yln, wqkv_p, qkv_b, qb, kb, vb, nullptr, nullptr);

  // ---- windowed attention ----
  attn_kernel<<<NWIN, 256, 0, stream>>>(qb, kb, vb, ao);

  // ---- output projection + window reverse + residual ----
  gemm_wmma<256, 256, EPI_PROJ><<<dim3(R_TOK / 64, 1), 256, 0, stream>>>(
      ao, wproj_p, proj_b, nullptr, nullptr, nullptr, x1, x);

  // ---- LN2 (no permutation) ----
  ln_kernel<false><<<R_TOK, 256, 0, stream>>>(x1, ln2_g, ln2_b, yln);

  // ---- MLP up + exact GELU ----
  gemm_wmma<1024, 256, EPI_GELU><<<dim3(R_TOK / 64, 4), 256, 0, stream>>>(
      yln, w1_p, mlp_b1, mh, nullptr, nullptr, nullptr, nullptr);

  // ---- MLP down + residual -> d_out (f32) ----
  gemm_wmma<256, 1024, EPI_OUT><<<dim3(R_TOK / 64, 1), 256, 0, stream>>>(
      mh, w2_p, mlp_b2, nullptr, nullptr, nullptr, (float*)d_out, x1);
}
